// SimpleRNN_85332410237199
// MI455X (gfx1250) — compile-verified, matrix operand order checked by reading, not run on hardware
//
#include <hip/hip_runtime.h>
#include <hip/hip_bf16.h>
#include <math.h>

// ---------------------------------------------------------------------------
// Problem constants: B=16, T=4096, H=1024, K=V=O=128, P=3K+V=512
// ---------------------------------------------------------------------------
#define BB 16
#define TT 4096
#define HH 1024
#define KK 128
#define VV 128
#define OO 128
#define PP 512
#define NROWS (BB * TT)

// Recurrence LDS layout (floats), padded strides to break bank conflicts:
//  s_state: [128][132]            @ 0          (16896 f)
//  chunk:   [16][516]  (q|k|g|v)  @ 16896      (8256 f)   cols: q 0,k 128,g 256,v 384
//  khat:    [128][17]             @ 25152      (2176 f)
//  Am:      [16][17]              @ 27328      (272 f)
#define SST   132
#define CLD   516
#define CH_OFF  16896
#define KH_OFF  25152
#define AM_OFF  27328
#define SMEM_F  27600   // total floats (110400 B)

typedef __attribute__((ext_vector_type(16))) __bf16 v16bf;
typedef __attribute__((ext_vector_type(8)))  __bf16 v8bf;
typedef __attribute__((ext_vector_type(4)))  __bf16 v4bf;
typedef __attribute__((ext_vector_type(8)))  float  v8f;
typedef __attribute__((ext_vector_type(2)))  float  v2f;

// ---------------------------------------------------------------------------
// WMMA wrappers
// ---------------------------------------------------------------------------
__device__ __forceinline__ v8f wmma_bf16(v8f c, v16bf a, v16bf b) {
    return __builtin_amdgcn_wmma_f32_16x16x32_bf16(false, a, false, b,
                                                   (short)0, c, false, false);
}
__device__ __forceinline__ v8f wmma_f32(v8f c, v2f a, v2f b) {
    return __builtin_amdgcn_wmma_f32_16x16x4_f32(false, a, false, b,
                                                 (short)0, c, false, false);
}

// 16-bit A/B fragment from row-major bf16 LDS tile, stride 40 (80B rows ->
// 16B-aligned groups, 20-bank rotation). ISA layout: lanes 0-15 hold K
// 0..7 & 16..23, lanes 16-31 hold K 8..15 & 24..31 -> two b128 loads.
__device__ __forceinline__ v16bf load_frag16(const __bf16* lds, int row, int hf) {
    const __bf16* p = lds + row * 40 + hf * 8;
    v8bf lo = *(const v8bf*)(p);
    v8bf hi = *(const v8bf*)(p + 16);
    return __builtin_shufflevector(lo, hi, 0, 1, 2, 3, 4, 5, 6, 7,
                                   8, 9, 10, 11, 12, 13, 14, 15);
}

__device__ __forceinline__ float sigmoidf_(float x) {
    return 1.0f / (1.0f + __expf(-x));
}

// ---------------------------------------------------------------------------
// Kernel 1: qkgv = act( X(65536x1024) @ Wp^T + b ), bf16 WMMA, fp32 out.
// Block tile 128x128, 8 waves (wave tile 32x64 = 2x4 subtiles).
// Each N-block covers exactly one segment -> activation choice is uniform.
// ---------------------------------------------------------------------------
__global__ void proj_gemm_kernel(const float* __restrict__ X,
                                 const float* __restrict__ Wp,
                                 const float* __restrict__ bias,
                                 float* __restrict__ out) {
    __shared__ __attribute__((aligned(16))) __bf16 lx[128 * 40];
    __shared__ __attribute__((aligned(16))) __bf16 lw[128 * 40];

    const int tid  = threadIdx.x;
    const int lane = tid & 31;
    const int wave = tid >> 5;
    const int hf   = lane >> 4;
    const int r    = lane & 15;
    const int mblk = blockIdx.x * 128;
    const int nblk = blockIdx.y * 128;
    const int wm   = (wave >> 1) * 32;
    const int wn   = (wave & 1) * 64;

    const v8f vzero = {0.f, 0.f, 0.f, 0.f, 0.f, 0.f, 0.f, 0.f};
    v8f acc[2][4];
#pragma unroll
    for (int i = 0; i < 2; ++i)
#pragma unroll
        for (int j = 0; j < 4; ++j) acc[i][j] = vzero;

    for (int kb = 0; kb < HH; kb += 32) {
        // stage 128x32 fp32 -> bf16 (float4 loads, packed b64 LDS stores)
#pragma unroll
        for (int e = 0; e < 4; ++e) {
            int gid = e * 256 + tid;          // 0..1023 float4 groups
            int row = gid >> 3;
            int col = (gid & 7) * 4;
            float4 xv = *(const float4*)&X [(size_t)(mblk + row) * HH + kb + col];
            float4 wv = *(const float4*)&Wp[(size_t)(nblk + row) * HH + kb + col];
            v4bf xp = {(__bf16)xv.x, (__bf16)xv.y, (__bf16)xv.z, (__bf16)xv.w};
            v4bf wp = {(__bf16)wv.x, (__bf16)wv.y, (__bf16)wv.z, (__bf16)wv.w};
            *(v4bf*)&lx[row * 40 + col] = xp;
            *(v4bf*)&lw[row * 40 + col] = wp;
        }
        // prefetch next K tile (global_prefetch_b8)
        if (kb + 32 < HH) {
            int prow = tid >> 1, pcol = (tid & 1) * 16;
            __builtin_prefetch(&X [(size_t)(mblk + prow) * HH + kb + 32 + pcol], 0, 3);
            __builtin_prefetch(&Wp[(size_t)(nblk + prow) * HH + kb + 32 + pcol], 0, 3);
        }
        __syncthreads();

        v16bf afr[2], bfr[4];
#pragma unroll
        for (int i = 0; i < 2; ++i) afr[i] = load_frag16(lx, wm + 16 * i + r, hf);
#pragma unroll
        for (int j = 0; j < 4; ++j) bfr[j] = load_frag16(lw, wn + 16 * j + r, hf);
#pragma unroll
        for (int i = 0; i < 2; ++i)
#pragma unroll
            for (int j = 0; j < 4; ++j)
                acc[i][j] = wmma_bf16(acc[i][j], afr[i], bfr[j]);
        __syncthreads();
    }

    // epilogue: bias + block-uniform activation
    const int seg = blockIdx.y;   // 0:q 1:k 2:g 3:v (uniform across block)
#pragma unroll
    for (int i = 0; i < 2; ++i) {
#pragma unroll
        for (int j = 0; j < 4; ++j) {
            const int col = nblk + wn + 16 * j + r;
            const float bv = bias[col];
#pragma unroll
            for (int e = 0; e < 8; ++e) {
                const int row = mblk + wm + 16 * i + e + 8 * hf;
                float v = acc[i][j][e] + bv;
                if (seg == 1 || seg == 2) v = sigmoidf_(v);
                else if (seg == 3)        v = tanhf(v);
                out[(size_t)row * PP + col] = v;
            }
        }
    }
}

// ---------------------------------------------------------------------------
// Kernel 2: chunked gated fast-weight recurrence + fused output projection.
// 16 blocks (one per batch), 256 thr = 8 waves; wave w owns state cols
// [16w,16w+16). Chunk staging uses GLOBAL_LOAD_ASYNC_TO_LDS_B128 (ASYNCcnt).
//   a=cumprod(g); Qh=q*a; Kh=k/a; Khat[k][s]=Kh[s][k]*a15[k]
//   Am=tril(Qh Kh^T); o=Am Vc + Qh S0; S=diag(a15)S0 + Khat Vc
//   out = o @ Wout^T + bout   (V_WMMA_F32_16X16X4_F32 throughout)
// ---------------------------------------------------------------------------
__global__ void recurrence_kernel(const float* __restrict__ qkgv,
                                  const float* __restrict__ Wout,
                                  const float* __restrict__ bout,
                                  float* __restrict__ out,
                                  float* __restrict__ cfinal) {
    extern __shared__ __attribute__((aligned(16))) float sm[];
    float* s_state = sm;            // [128][132]
    float* ch      = sm + CH_OFF;   // [16][516]: q|k(->Kh,o)|g(->a)|v
    float* khat    = sm + KH_OFF;   // [128][17]
    float* Am      = sm + AM_OFF;   // [16][17]

    const int tid  = threadIdx.x;
    const int lane = tid & 31;
    const int wave = tid >> 5;
    const int hf   = lane >> 4;
    const int r    = lane & 15;
    const int b    = blockIdx.x;
    const int n0   = wave * 16;
    const v8f vzero = {0.f, 0.f, 0.f, 0.f, 0.f, 0.f, 0.f, 0.f};

    for (int e = tid; e < 128 * SST; e += 256) s_state[e] = 0.f;
    __syncthreads();

    for (int tc = 0; tc < TT / 16; ++tc) {
        // ---- 1. async copy chunk (16x512 contiguous fp32) into padded LDS --
        const size_t base = ((size_t)b * TT + (size_t)tc * 16) * PP;
#pragma unroll
        for (int e = 0; e < 8; ++e) {
            int idx4 = (e * 256 + tid) * 4;          // 0..8188 step 4
            int row  = idx4 >> 9;
            int col  = idx4 & 511;
            unsigned ldsoff = (unsigned)(CH_OFF + row * CLD + col) * 4u;
            unsigned long long gaddr =
                (unsigned long long)(uintptr_t)(qkgv + base + idx4);
            asm volatile("global_load_async_to_lds_b128 %0, %1, off"
                         :: "v"(ldsoff), "v"(gaddr) : "memory");
        }
        asm volatile("s_wait_asynccnt 0x0" ::: "memory");
        __syncthreads();

        // ---- 2. cumprod + Qh/Kh/Khat (threads 0..127, one per k) ----
        if (tid < 128) {
            const int k = tid;
            float a = 1.f;
#pragma unroll
            for (int t = 0; t < 16; ++t) {
                a *= ch[t * CLD + 256 + k];
                a  = fmaxf(a, 1e-30f);
                ch[t * CLD + 256 + k] = a;           // a_t
                ch[t * CLD + k]      *= a;           // Qh
                ch[t * CLD + 128 + k] *= 1.0f / a;   // Kh
            }
            const float a15 = a;
#pragma unroll
            for (int t = 0; t < 16; ++t)
                khat[k * 17 + t] = ch[t * CLD + 128 + k] * a15;
        }
        __syncthreads();

        // ---- 3. wave 0: Am = tril( Qh @ Kh^T ), K=128 ----
        if (wave == 0) {
            v8f ac = vzero;
            for (int kb = 0; kb < 128; kb += 4) {
                v2f af, bf;
                af.x = ch[r * CLD + kb + 2 * hf];
                af.y = ch[r * CLD + kb + 1 + 2 * hf];
                bf.x = ch[r * CLD + 128 + kb + 2 * hf];
                bf.y = ch[r * CLD + 128 + kb + 1 + 2 * hf];
                ac = wmma_f32(ac, af, bf);
            }
#pragma unroll
            for (int e = 0; e < 8; ++e) {
                int t = e + 8 * hf;
                Am[t * 17 + r] = (r <= t) ? ac[e] : 0.f;
            }
        }
        __syncthreads();

        // ---- 4a. o = Qh@S0 + Am@Vc ; state update (per-wave column slice) --
        v8f oc = vzero;
        for (int kb = 0; kb < 128; kb += 4) {        // inter: Qh @ S0
            v2f af, bf;
            af.x = ch[r * CLD + kb + 2 * hf];
            af.y = ch[r * CLD + kb + 1 + 2 * hf];
            bf.x = s_state[(kb + 2 * hf) * SST + n0 + r];
            bf.y = s_state[(kb + 1 + 2 * hf) * SST + n0 + r];
            oc = wmma_f32(oc, af, bf);
        }
#pragma unroll
        for (int sb = 0; sb < 16; sb += 4) {         // intra: Am @ Vc
            v2f af, bf;
            af.x = Am[r * 17 + sb + 2 * hf];
            af.y = Am[r * 17 + sb + 1 + 2 * hf];
            bf.x = ch[(sb + 2 * hf) * CLD + 384 + n0 + r];
            bf.y = ch[(sb + 1 + 2 * hf) * CLD + 384 + n0 + r];
            oc = wmma_f32(oc, af, bf);
        }
        // S = diag(a15) S0 + Khat @ Vc
#pragma unroll
        for (int mt = 0; mt < 8; ++mt) {
            const int m0 = mt * 16;
            v8f st;
#pragma unroll
            for (int e = 0; e < 8; ++e) {
                int m = m0 + e + 8 * hf;
                st[e] = s_state[m * SST + n0 + r] * ch[15 * CLD + 256 + m];
            }
#pragma unroll
            for (int sb = 0; sb < 16; sb += 4) {
                v2f af, bf;
                af.x = khat[(m0 + r) * 17 + sb + 2 * hf];
                af.y = khat[(m0 + r) * 17 + sb + 1 + 2 * hf];
                bf.x = ch[(sb + 2 * hf) * CLD + 384 + n0 + r];
                bf.y = ch[(sb + 1 + 2 * hf) * CLD + 384 + n0 + r];
                st = wmma_f32(st, af, bf);
            }
#pragma unroll
            for (int e = 0; e < 8; ++e) {
                int m = m0 + e + 8 * hf;
                s_state[m * SST + n0 + r] = st[e];
            }
        }
        // stage o-chunk into LDS (reuse Kh columns) for fused out-projection
#pragma unroll
        for (int e = 0; e < 8; ++e) {
            int t = e + 8 * hf;
            ch[t * CLD + 128 + n0 + r] = oc[e];
        }
        __syncthreads();

        // ---- 4b. out = o_chunk @ Wout^T + bout ----
        v8f dc = vzero;
        for (int kb = 0; kb < 128; kb += 4) {
            v2f af, bf;
            af.x = ch[r * CLD + 128 + kb + 2 * hf];
            af.y = ch[r * CLD + 128 + kb + 1 + 2 * hf];
            bf.x = Wout[(size_t)(n0 + r) * VV + kb + 2 * hf];
            bf.y = Wout[(size_t)(n0 + r) * VV + kb + 1 + 2 * hf];
            dc = wmma_f32(dc, af, bf);
        }
        const float bo = bout[n0 + r];
        const size_t orow = (size_t)b * TT + (size_t)tc * 16;
#pragma unroll
        for (int e = 0; e < 8; ++e) {
            int t = e + 8 * hf;
            out[(orow + t) * OO + n0 + r] = dc[e] + bo;
        }
        __syncthreads();   // protect chunk buffers before next async load
    }

    // ---- c_final: (B,K,V) ----
    for (int e = tid; e < KK * VV; e += 256) {
        int k = e >> 7, v = e & 127;
        cfinal[(size_t)b * (KK * VV) + e] = s_state[k * SST + v];
    }
}

// ---------------------------------------------------------------------------
// Launch
// ---------------------------------------------------------------------------
extern "C" void kernel_launch(void* const* d_in, const int* in_sizes, int n_in,
                              void* d_out, int out_size, void* d_ws, size_t ws_size,
                              hipStream_t stream) {
    const float* X  = (const float*)d_in[0];
    const float* Wp = (const float*)d_in[1];
    const float* bp = (const float*)d_in[2];
    const float* Wo = (const float*)d_in[3];
    const float* bo = (const float*)d_in[4];

    float* out  = (float*)d_out;
    float* cfin = out + (size_t)BB * TT * OO;
    float* qkgv = (float*)d_ws;   // (B*T, 512) fp32

    dim3 g1(NROWS / 128, PP / 128);   // 512 x 4
    proj_gemm_kernel<<<g1, 256, 0, stream>>>(X, Wp, bp, qkgv);

    const size_t smem = (size_t)SMEM_F * sizeof(float);
    (void)hipFuncSetAttribute((const void*)recurrence_kernel,
                              hipFuncAttributeMaxDynamicSharedMemorySize, (int)smem);
    recurrence_kernel<<<BB, 256, smem, stream>>>(qkgv, Wo, bo, out, cfin);
}
